// HomoGNN_55559696941682
// MI455X (gfx1250) — matrix operand order repaired and checked by reading, not compiled, not run on hardware
//
#include <hip/hip_runtime.h>
#include <hip/hip_bf16.h>
#include <math.h>

// ---------------------------------------------------------------------------
// HomoGNN (3-layer GCN) for MI455X / gfx1250, wave32 + WMMA bf16.
//
// gcn_conv(x,W) = A_norm @ (x@W) + b  with A_norm linear in node space, so
// aggregation commutes with the weight GEMM. We always aggregate in the
// 16-dim hidden space (E*16 float atomics per layer instead of E*128 or
// E*300), then do the GEMMs with v_wmma_f32_16x16x32_bf16.
// ---------------------------------------------------------------------------

typedef __attribute__((ext_vector_type(16))) __bf16 v16bf;
typedef __attribute__((ext_vector_type(8)))  float  v8f;

#define IN_DIM  128
#define HD      16
#define OUT_DIM 300

// ---------------- degree / norm ----------------
__global__ void init_deg_k(float* deg, int n) {
  int i = blockIdx.x * blockDim.x + threadIdx.x;
  if (i < n) deg[i] = 1.0f;                      // +1 self loop
}
__global__ void accum_deg_k(const int* __restrict__ dst, float* deg, int e) {
  int i = blockIdx.x * blockDim.x + threadIdx.x;
  if (i < e) atomicAdd(&deg[dst[i]], 1.0f);
}
__global__ void dinv_k(const float* __restrict__ deg, float* dinv, int n) {
  int i = blockIdx.x * blockDim.x + threadIdx.x;
  if (i < n) dinv[i] = rsqrtf(deg[i]);
}

// ---------------- layer 1 GEMM: X[N,128] @ {W1,Wl1}[128,16] ----------------
// One wave per 16-row tile; shared A fragment, two accumulators.
__global__ void gemm_x_dual_k(const float* __restrict__ X,
                              const float* __restrict__ W,
                              const float* __restrict__ Wl,
                              float* __restrict__ Hout,
                              float* __restrict__ Sout, int n) {
  const int lane = threadIdx.x & 31;
  const int tile = blockIdx.x * (blockDim.x >> 5) + (threadIdx.x >> 5);
  const int tiles = (n + 15) >> 4;
  if (tile >= tiles) return;                     // uniform per wave
  const int m = lane & 15, g = lane >> 4;
  const int row = tile * 16 + m;

  v8f accH = {}; v8f accS = {};
  for (int kb = 0; kb < IN_DIM; kb += 32) {
    v16bf a, bH, bS;
#pragma unroll
    for (int j = 0; j < 8; ++j) {
      // A (16x32 bf16) lane layout: VGPR j holds K = kp,kp+1
      const int kp = g * 16 + 2 * j;
      float2 xv = *(const float2*)(X + (size_t)row * IN_DIM + kb + kp);
      a[2 * j]     = (__bf16)xv.x;
      a[2 * j + 1] = (__bf16)xv.y;
      // B (32x16 bf16): lanes 0-15 K=0..15, lanes 16-31 K=16..31, N=m
      const int kB = g * 16 + 2 * j;
      bH[2 * j]     = (__bf16)W [(kb + kB)     * HD + m];
      bH[2 * j + 1] = (__bf16)W [(kb + kB + 1) * HD + m];
      bS[2 * j]     = (__bf16)Wl[(kb + kB)     * HD + m];
      bS[2 * j + 1] = (__bf16)Wl[(kb + kB + 1) * HD + m];
    }
    accH = __builtin_amdgcn_wmma_f32_16x16x32_bf16(false, a, false, bH,
                                                   (short)0, accH, false, false);
    accS = __builtin_amdgcn_wmma_f32_16x16x32_bf16(false, a, false, bS,
                                                   (short)0, accS, false, false);
  }
#pragma unroll
  for (int v = 0; v < 8; ++v) {                  // C/D: VGPR v -> M=v+8g, N=m
    const int r = tile * 16 + v + g * 8;
    Hout[(size_t)r * HD + m] = accH[v];
    Sout[(size_t)r * HD + m] = accS[v];
  }
}

// ---------------- hidden GEMM: Hin[N,16] @ {W,Wl}[16,16], K padded to 32 ----
__global__ void gemm_h_dual_k(const float* __restrict__ Hin,
                              const float* __restrict__ W,
                              const float* __restrict__ Wl,
                              float* __restrict__ Hout,
                              float* __restrict__ Sout, int n) {
  const int lane = threadIdx.x & 31;
  const int tile = blockIdx.x * (blockDim.x >> 5) + (threadIdx.x >> 5);
  const int tiles = (n + 15) >> 4;
  if (tile >= tiles) return;
  const int m = lane & 15, g = lane >> 4;
  const int row = tile * 16 + m;

  v16bf a, bH, bS;
#pragma unroll
  for (int j = 0; j < 8; ++j) {
    if (g == 0) {                                // K = 2j in [0,16)
      const int kp = 2 * j;
      float2 hv = *(const float2*)(Hin + (size_t)row * HD + kp);
      a[2 * j]     = (__bf16)hv.x;
      a[2 * j + 1] = (__bf16)hv.y;
    } else {                                     // K >= 16 : zero pad
      a[2 * j] = (__bf16)0.0f; a[2 * j + 1] = (__bf16)0.0f;
    }
    const int kB = 2 * j;                        // lanes g==1 carry K>=16 -> 0
    bH[2 * j]     = (g == 0) ? (__bf16)W [kB       * HD + m] : (__bf16)0.0f;
    bH[2 * j + 1] = (g == 0) ? (__bf16)W [(kB + 1) * HD + m] : (__bf16)0.0f;
    bS[2 * j]     = (g == 0) ? (__bf16)Wl[kB       * HD + m] : (__bf16)0.0f;
    bS[2 * j + 1] = (g == 0) ? (__bf16)Wl[(kB + 1) * HD + m] : (__bf16)0.0f;
  }
  v8f accH = {}; v8f accS = {};
  accH = __builtin_amdgcn_wmma_f32_16x16x32_bf16(false, a, false, bH,
                                                 (short)0, accH, false, false);
  accS = __builtin_amdgcn_wmma_f32_16x16x32_bf16(false, a, false, bS,
                                                 (short)0, accS, false, false);
#pragma unroll
  for (int v = 0; v < 8; ++v) {
    const int r = tile * 16 + v + g * 8;
    Hout[(size_t)r * HD + m] = accH[v];
    Sout[(size_t)r * HD + m] = accS[v];
  }
}

// ---------------- aggregation (16-dim space) ----------------
__global__ void self_init16_k(const float* __restrict__ F,
                              const float* __restrict__ dinv,
                              float* __restrict__ AGG, int n) {
  int i = blockIdx.x * blockDim.x + threadIdx.x;
  if (i < n * HD) {
    int node = i >> 4;
    float dv = dinv[node];
    AGG[i] = F[i] * dv * dv;                     // self-loop message
  }
}
__global__ void edge_scatter16_k(const float* __restrict__ F,
                                 const int* __restrict__ src,
                                 const int* __restrict__ dst,
                                 const float* __restrict__ dinv,
                                 float* __restrict__ AGG, int e) {
  int i = blockIdx.x * blockDim.x + threadIdx.x; // 16 lanes per edge
  int ed = i >> 4, d = i & 15;
  if (ed < e) {
    int s = src[ed], t = dst[ed];
    float norm = dinv[s] * dinv[t];
    atomicAdd(&AGG[(size_t)t * HD + d], F[(size_t)s * HD + d] * norm);
  }
}

// ---------------- relu(conv)+skip combine ----------------
__global__ void combine16_k(const float* __restrict__ AGG,
                            const float* __restrict__ b,
                            const float* __restrict__ S,
                            const float* __restrict__ bl,
                            float* __restrict__ Hout, int n) {
  int i = blockIdx.x * blockDim.x + threadIdx.x;
  if (i < n * HD) {
    int d = i & 15;
    Hout[i] = fmaxf(AGG[i] + b[d], 0.0f) + S[i] + bl[d];
  }
}

// ---------------- output GEMM: AGG[N,16] @ Wo[16,300] + bo ----------------
__global__ void gemm_out_k(const float* __restrict__ AGG,
                           const float* __restrict__ Wo,
                           const float* __restrict__ bo,
                           float* __restrict__ OUT, int n) {
  const int lane = threadIdx.x & 31;
  const int tile = blockIdx.x * (blockDim.x >> 5) + (threadIdx.x >> 5);
  const int tiles = (n + 15) >> 4;
  if (tile >= tiles) return;
  const int m = lane & 15, g = lane >> 4;
  const int row = tile * 16 + m;

  v16bf a;                                       // A fragment loaded once
#pragma unroll
  for (int j = 0; j < 8; ++j) {
    if (g == 0) {
      const int kp = 2 * j;
      float2 hv = *(const float2*)(AGG + (size_t)row * HD + kp);
      a[2 * j]     = (__bf16)hv.x;
      a[2 * j + 1] = (__bf16)hv.y;
    } else {
      a[2 * j] = (__bf16)0.0f; a[2 * j + 1] = (__bf16)0.0f;
    }
  }
  const int ctiles = (OUT_DIM + 15) / 16;        // 19 column tiles
  for (int ct = 0; ct < ctiles; ++ct) {
    const int col = ct * 16 + m;
    const bool ok = (g == 0) && (col < OUT_DIM);
    v16bf b;
#pragma unroll
    for (int j = 0; j < 8; ++j) {
      const int kB = 2 * j;
      b[2 * j]     = ok ? (__bf16)Wo[(size_t)kB       * OUT_DIM + col] : (__bf16)0.0f;
      b[2 * j + 1] = ok ? (__bf16)Wo[(size_t)(kB + 1) * OUT_DIM + col] : (__bf16)0.0f;
    }
    v8f c = {};
    c = __builtin_amdgcn_wmma_f32_16x16x32_bf16(false, a, false, b,
                                                (short)0, c, false, false);
    if (col < OUT_DIM) {
      const float bias = bo[col];
#pragma unroll
      for (int v = 0; v < 8; ++v) {
        const int r = tile * 16 + v + g * 8;
        OUT[(size_t)r * OUT_DIM + col] = c[v] + bias;
      }
    }
  }
}

// ---------------- row-wise log_softmax, one wave32 per row ----------------
__global__ void log_softmax_k(float* __restrict__ OUT, int n) {
  const int lane = threadIdx.x & 31;
  const int row  = blockIdx.x * (blockDim.x >> 5) + (threadIdx.x >> 5);
  if (row >= n) return;
  float* r = OUT + (size_t)row * OUT_DIM;
  float mx = -3.402823466e38f;
  for (int c = lane; c < OUT_DIM; c += 32) mx = fmaxf(mx, r[c]);
#pragma unroll
  for (int o = 16; o > 0; o >>= 1) mx = fmaxf(mx, __shfl_xor(mx, o, 32));
  float s = 0.0f;
  for (int c = lane; c < OUT_DIM; c += 32) s += __expf(r[c] - mx);
#pragma unroll
  for (int o = 16; o > 0; o >>= 1) s += __shfl_xor(s, o, 32);
  const float lse = __logf(s) + mx;
  for (int c = lane; c < OUT_DIM; c += 32) r[c] = r[c] - lse;
}

// ---------------------------------------------------------------------------
extern "C" void kernel_launch(void* const* d_in, const int* in_sizes, int n_in,
                              void* d_out, int out_size, void* d_ws, size_t ws_size,
                              hipStream_t stream) {
  const float* x   = (const float*)d_in[0];
  const int*   ei  = (const int*)  d_in[1];
  const float* W1  = (const float*)d_in[2];
  const float* b1  = (const float*)d_in[3];
  const float* Wl1 = (const float*)d_in[4];
  const float* bl1 = (const float*)d_in[5];
  const float* W2  = (const float*)d_in[6];
  const float* b2  = (const float*)d_in[7];
  const float* Wl2 = (const float*)d_in[8];
  const float* bl2 = (const float*)d_in[9];
  const float* Wo  = (const float*)d_in[10];
  const float* bo  = (const float*)d_in[11];
  float* out = (float*)d_out;

  const int N = in_sizes[0] / IN_DIM;
  const int E = in_sizes[1] / 2;
  const int* src = ei;
  const int* dst = ei + E;

  float* ws   = (float*)d_ws;
  float* deg  = ws;  ws += N;
  float* dinv = ws;  ws += N;
  float* Hb   = ws;  ws += (size_t)N * HD;   // pre-aggregation GEMM result
  float* Sb   = ws;  ws += (size_t)N * HD;   // skip-connection GEMM result
  float* AGG  = ws;  ws += (size_t)N * HD;   // aggregation buffer
  float* h1   = ws;  ws += (size_t)N * HD;
  float* h2   = ws;  ws += (size_t)N * HD;

  const int T   = 256;
  const int gN  = (N + T - 1) / T;
  const int gE  = (E + T - 1) / T;
  const int gNH = (N * HD + T - 1) / T;
  const int gEH = (E * HD + T - 1) / T;
  const int tiles = (N + 15) / 16;
  const int gTile = (tiles + 7) / 8;           // 8 waves / block

  // degrees + symmetric norm
  init_deg_k <<<gN, T, 0, stream>>>(deg, N);
  accum_deg_k<<<gE, T, 0, stream>>>(dst, deg, E);
  dinv_k     <<<gN, T, 0, stream>>>(deg, dinv, N);

  // layer 1: GEMM (128->16) first, then aggregate in 16-dim space
  gemm_x_dual_k  <<<gTile, T, 0, stream>>>(x, W1, Wl1, Hb, Sb, N);
  self_init16_k  <<<gNH, T, 0, stream>>>(Hb, dinv, AGG, N);
  edge_scatter16_k<<<gEH, T, 0, stream>>>(Hb, src, dst, dinv, AGG, E);
  combine16_k    <<<gNH, T, 0, stream>>>(AGG, b1, Sb, bl1, h1, N);

  // layer 2
  gemm_h_dual_k  <<<gTile, T, 0, stream>>>(h1, W2, Wl2, Hb, Sb, N);
  self_init16_k  <<<gNH, T, 0, stream>>>(Hb, dinv, AGG, N);
  edge_scatter16_k<<<gEH, T, 0, stream>>>(Hb, src, dst, dinv, AGG, E);
  combine16_k    <<<gNH, T, 0, stream>>>(AGG, b2, Sb, bl2, h2, N);

  // output: aggregate FIRST in 16-dim space, then GEMM 16->300 (+bias)
  self_init16_k  <<<gNH, T, 0, stream>>>(h2, dinv, AGG, N);
  edge_scatter16_k<<<gEH, T, 0, stream>>>(h2, src, dst, dinv, AGG, E);
  gemm_out_k     <<<gTile, T, 0, stream>>>(AGG, Wo, bo, out, N);

  // log_softmax in place on d_out
  log_softmax_k  <<<(N + 7) / 8, T, 0, stream>>>(out, N);
}
